// SchNetInteraction_39840116638269
// MI455X (gfx1250) — matrix operand order repaired and checked by reading, not compiled, not run on hardware
//
#include <hip/hip_runtime.h>

typedef unsigned short u16;
typedef unsigned int   u32;
typedef __attribute__((ext_vector_type(16))) __bf16 v16bf;
typedef __attribute__((ext_vector_type(8)))  float  v8f;

#define DIMN 64
#define HH   2
#define HD   128   // H*DIM
#define HID  128
#define NG   64
#define RCUT 5.0f
#define LN2F 0.69314718055994531f

union Frag { v16bf v; uint4 q[2]; };

__device__ __forceinline__ u16 f2bf(float f) {
  u32 u = __float_as_uint(f);
  u32 r = u + 0x7fffu + ((u >> 16) & 1u);   // round-to-nearest-even
  return (u16)(r >> 16);
}

__device__ __forceinline__ float sspf(float x) {
  // softplus(x) - ln2, numerically stable
  return fmaxf(x, 0.f) + __logf(1.f + __expf(-fabsf(x))) - LN2F;
}

// order-preserving float -> uint key (for atomicMax-based segment max)
__device__ __forceinline__ u32 fkey(float f) {
  u32 u = __float_as_uint(f);
  return (u & 0x80000000u) ? ~u : (u | 0x80000000u);
}
__device__ __forceinline__ float funkey(u32 k) {
  u32 u = (k & 0x80000000u) ? (k & 0x7fffffffu) : ~k;
  return __uint_as_float(u);
}

// A-matrix (16x32 bf16, MxK), wave32 layout: row M = lane&15; half 0 holds
// K=[kb..kb+7] then [kb+16..kb+23]; half 1 holds [kb+8..+15] then [kb+24..+31].
__device__ __forceinline__ Frag loadA(const u16* __restrict__ tile, int stride, int kb) {
  int l = threadIdx.x & 31;
  int M = l & 15, half = l >> 4;
  Frag a;
  const u16* p = tile + M * stride + kb + half * 8;
  a.q[0] = *(const uint4*)(p);
  a.q[1] = *(const uint4*)(p + 16);
  return a;
}

// B-matrix (32x16 bf16, KxN): column N = lane&15; half 0 holds K=[kb..kb+7] then
// [kb+16..+23]; half 1 holds [kb+8..+15] then [kb+24..+31]. Weights stored transposed [Nout][K] in LDS.
__device__ __forceinline__ Frag loadB(const u16* __restrict__ wT, int Kdim, int nb, int kb) {
  int l = threadIdx.x & 31;
  int N = nb + (l & 15), half = l >> 4;
  Frag b;
  const u16* p = wT + N * Kdim + kb + half * 8;
  b.q[0] = *(const uint4*)(p);
  b.q[1] = *(const uint4*)(p + 16);
  return b;
}

__device__ __forceinline__ v8f bfwmma(const Frag& a, const Frag& b, v8f c) {
  return __builtin_amdgcn_wmma_f32_16x16x32_bf16(false, a.v, false, b.v, (short)0, c,
                                                 false, false);
}

// ---------------- Kernel 1: feat = x @ fc_w  (N x 64) @ (64 x 128) ----------------
// 32-node tiles; 8 waves each own a 16-wide N-tile; B fragment shared by 2 row-blocks.
__global__ void __launch_bounds__(256)
feat_kernel(const float* __restrict__ x, const float* __restrict__ fcw,
            float* __restrict__ feat, int N) {
  __shared__ __align__(16) u16 sW[HD * DIMN];   // fc_w^T [128][64] 16KB
  __shared__ __align__(16) u16 sX[32 * DIMN];   // 4KB
  for (int idx = threadIdx.x; idx < DIMN * HD; idx += blockDim.x) {
    int k = idx / HD, n = idx % HD;
    sW[n * DIMN + k] = f2bf(fcw[idx]);
  }
  int nt = (N + 31) / 32;
  int lane = threadIdx.x & 31, wv = threadIdx.x >> 5, half = lane >> 4;
  for (int t = blockIdx.x; t < nt; t += gridDim.x) {
    int n0 = t * 32;
    bool full = (n0 + 32 <= N);
    __syncthreads();
    if (full) {
      for (int idx = threadIdx.x; idx < 32 * DIMN; idx += blockDim.x)
        sX[idx] = f2bf(x[(size_t)n0 * DIMN + idx]);
    } else {
      for (int idx = threadIdx.x; idx < 32 * DIMN; idx += blockDim.x) {
        int r = idx / DIMN, c = idx % DIMN;
        int node = n0 + r;
        sX[idx] = f2bf(node < N ? x[(size_t)node * DIMN + c] : 0.f);
      }
    }
    __syncthreads();
    v8f acc0 = {}, acc1 = {};
#pragma unroll
    for (int kb = 0; kb < DIMN; kb += 32) {
      Frag b = loadB(sW, DIMN, wv * 16, kb);
      Frag a0 = loadA(sX, DIMN, kb);
      Frag a1 = loadA(sX + 16 * DIMN, DIMN, kb);
      acc0 = bfwmma(a0, b, acc0);
      acc1 = bfwmma(a1, b, acc1);
    }
    int c = wv * 16 + (lane & 15);
    if (full) {
      // one base address per lane; rows become immediate byte offsets (stride 512B)
      float* p0 = feat + (size_t)(n0 + 8 * half) * HD + c;
#pragma unroll
      for (int j = 0; j < 8; j++) {
        p0[j * HD] = acc0[j];
        p0[(j + 16) * HD] = acc1[j];
      }
    } else {
#pragma unroll
      for (int j = 0; j < 8; j++) {
        int r = j + 8 * half;
        if (n0 + r < N) feat[(size_t)(n0 + r) * HD + c] = acc0[j];
        if (n0 + r + 16 < N) feat[(size_t)(n0 + r + 16) * HD + c] = acc1[j];
      }
    }
  }
}

// ---------------- Kernel 2: el/er attention reductions (one wave per node) ----------------
__global__ void __launch_bounds__(256)
attn_kernel(const float* __restrict__ feat, const float* __restrict__ al,
            const float* __restrict__ ar, float* __restrict__ el,
            float* __restrict__ er, int N) {
  int wv = (blockIdx.x * blockDim.x + threadIdx.x) >> 5;
  int lane = threadIdx.x & 31;
  if (wv >= N) return;
  const float* f = feat + (size_t)wv * HD;
  float v0 = f[lane], v1 = f[lane + 32];          // head 0
  float w0 = f[64 + lane], w1 = f[96 + lane];     // head 1
  float sl0 = v0 * al[lane] + v1 * al[lane + 32];
  float sr0 = v0 * ar[lane] + v1 * ar[lane + 32];
  float sl1 = w0 * al[64 + lane] + w1 * al[96 + lane];
  float sr1 = w0 * ar[64 + lane] + w1 * ar[96 + lane];
#pragma unroll
  for (int off = 16; off; off >>= 1) {
    sl0 += __shfl_xor(sl0, off, 32);
    sr0 += __shfl_xor(sr0, off, 32);
    sl1 += __shfl_xor(sl1, off, 32);
    sr1 += __shfl_xor(sr1, off, 32);
  }
  if (lane == 0) {
    el[wv * 2] = sl0; el[wv * 2 + 1] = sl1;
    er[wv * 2] = sr0; er[wv * 2 + 1] = sr1;
  }
}

// ---------------- Kernel 3: edge scores + segment max ----------------
__global__ void score_kernel(const int* __restrict__ src, const int* __restrict__ dst,
                             const float* __restrict__ el, const float* __restrict__ er,
                             float* __restrict__ esc, u32* __restrict__ mkey, int E) {
  int e = blockIdx.x * blockDim.x + threadIdx.x;
  if (e >= E) return;
  int s = src[e], d = dst[e];
#pragma unroll
  for (int h = 0; h < HH; h++) {
    float v = el[s * HH + h] + er[d * HH + h];
    v = v > 0.f ? v : 0.2f * v;   // leaky_relu 0.2
    esc[(size_t)e * HH + h] = v;
    atomicMax(&mkey[d * HH + h], fkey(v));
  }
}

// ---------------- Kernel 4: exp + segment sum ----------------
__global__ void exp_kernel(const int* __restrict__ dst, float* __restrict__ esc,
                           const u32* __restrict__ mkey, float* __restrict__ den, int E) {
  int e = blockIdx.x * blockDim.x + threadIdx.x;
  if (e >= E) return;
  int d = dst[e];
#pragma unroll
  for (int h = 0; h < HH; h++) {
    float m = funkey(mkey[d * HH + h]);
    if (!(fabsf(m) <= 3.0e38f)) m = 0.f;   // non-finite -> 0 (reference semantics)
    float ex = __expf(esc[(size_t)e * HH + h] - m);
    esc[(size_t)e * HH + h] = ex;   // overwrite in place
    unsafeAtomicAdd(&den[d * HH + h], ex);
  }
}

// ------- Kernel 5: fused filter-net + message + scatter-add (dominant kernel) -------
// 32-edge tiles; 8 waves each own a 16-wide N-tile of the 128 output channels.
__global__ void __launch_bounds__(256)
msg_kernel(const float* __restrict__ gs, const float* __restrict__ rij,
           const int* __restrict__ src, const int* __restrict__ dst,
           const float* __restrict__ w1, const float* __restrict__ b1,
           const float* __restrict__ w2, const float* __restrict__ b2,
           const float* __restrict__ feat, const float* __restrict__ exsc,
           const float* __restrict__ den, float* __restrict__ agg, int E) {
  __shared__ __align__(16) u16 sW1[HID * NG];    // filt_w1^T [128][64]  16KB
  __shared__ __align__(16) u16 sW2[HD * HID];    // filt_w2^T [128][128] 32KB
  __shared__ float sB1[HID], sB2[HD];
  __shared__ __align__(16) u16 sG[32 * NG];      // gs tile   4KB
  __shared__ __align__(16) u16 sH1[32 * HID];    // hidden    8KB
  __shared__ float sCoef[32 * HH];
  __shared__ int sS[32], sD[32];

  for (int idx = threadIdx.x; idx < NG * HID; idx += blockDim.x) {
    int k = idx / HID, n = idx % HID;
    sW1[n * NG + k] = f2bf(w1[idx]);
  }
  for (int idx = threadIdx.x; idx < HID * HD; idx += blockDim.x) {
    int k = idx / HD, n = idx % HD;
    sW2[n * HID + k] = f2bf(w2[idx]);
  }
  for (int idx = threadIdx.x; idx < HID; idx += blockDim.x) sB1[idx] = b1[idx];
  for (int idx = threadIdx.x; idx < HD; idx += blockDim.x) sB2[idx] = b2[idx];

  int nt = (E + 31) / 32;
  int lane = threadIdx.x & 31, wv = threadIdx.x >> 5, half = lane >> 4;
  for (int t = blockIdx.x; t < nt; t += gridDim.x) {
    int e0 = t * 32;
    bool full = (e0 + 32 <= E);
    __syncthreads();
    if (full) {
      for (int idx = threadIdx.x; idx < 32 * NG; idx += blockDim.x)
        sG[idx] = f2bf(gs[(size_t)e0 * NG + idx]);
    } else {
      for (int idx = threadIdx.x; idx < 32 * NG; idx += blockDim.x) {
        int r = idx / NG, c = idx % NG;
        int e = e0 + r;
        sG[idx] = f2bf(e < E ? gs[(size_t)e * NG + c] : 0.f);
      }
    }
    if (threadIdx.x < 32) {
      int r = threadIdx.x;
      int e = e0 + r;
      if (e < E) {
        int d = dst[e];
        sS[r] = src[e];
        sD[r] = d;
        float rc = rij[e];
        float cut = rc < RCUT ? 0.5f * (__cosf(3.14159265358979f * rc / RCUT) + 1.f) : 0.f;
#pragma unroll
        for (int h = 0; h < HH; h++) {
          float dn = fmaxf(den[d * HH + h], 1e-16f);
          sCoef[r * HH + h] = exsc[(size_t)e * HH + h] / dn * cut;
        }
      } else {
        sS[r] = 0; sD[r] = 0;
        sCoef[r * HH] = 0.f; sCoef[r * HH + 1] = 0.f;
      }
    }
    __syncthreads();
    // GEMM1: h1 = ssp(gs @ W1 + b1)
    v8f acc0 = {}, acc1 = {};
#pragma unroll
    for (int kb = 0; kb < NG; kb += 32) {
      Frag b = loadB(sW1, NG, wv * 16, kb);
      Frag a0 = loadA(sG, NG, kb);
      Frag a1 = loadA(sG + 16 * NG, NG, kb);
      acc0 = bfwmma(a0, b, acc0);
      acc1 = bfwmma(a1, b, acc1);
    }
    int c1 = wv * 16 + (lane & 15);
    float bias1 = sB1[c1];
    {
      u16* q = sH1 + (8 * half) * HID + c1;
#pragma unroll
      for (int j = 0; j < 8; j++) {
        q[j * HID] = f2bf(sspf(acc0[j] + bias1));
        q[(j + 16) * HID] = f2bf(sspf(acc1[j] + bias1));
      }
    }
    __syncthreads();
    // GEMM2: filt = h1 @ W2 + b2, then message + scatter-add
    v8f d0 = {}, d1 = {};
#pragma unroll
    for (int kb = 0; kb < HID; kb += 32) {
      Frag b = loadB(sW2, HID, wv * 16, kb);
      Frag a0 = loadA(sH1, HID, kb);
      Frag a1 = loadA(sH1 + 16 * HID, HID, kb);
      d0 = bfwmma(a0, b, d0);
      d1 = bfwmma(a1, b, d1);
    }
    int c = wv * 16 + (lane & 15);
    int h = c >> 6;
    float bias2 = sB2[c];
    if (full) {
#pragma unroll
      for (int j = 0; j < 8; j++) {
        int r = j + 8 * half;
        float f0 = d0[j] + bias2;
        float v0 = feat[(size_t)sS[r] * HD + c] * f0 * sCoef[r * HH + h];
        unsafeAtomicAdd(&agg[(size_t)sD[r] * HD + c], v0);
        int r1 = r + 16;
        float f1 = d1[j] + bias2;
        float v1 = feat[(size_t)sS[r1] * HD + c] * f1 * sCoef[r1 * HH + h];
        unsafeAtomicAdd(&agg[(size_t)sD[r1] * HD + c], v1);
      }
    } else {
#pragma unroll
      for (int j = 0; j < 8; j++) {
        int r = j + 8 * half;
        if (e0 + r < E) {
          float f0 = d0[j] + bias2;
          float v0 = feat[(size_t)sS[r] * HD + c] * f0 * sCoef[r * HH + h];
          unsafeAtomicAdd(&agg[(size_t)sD[r] * HD + c], v0);
        }
        int r1 = r + 16;
        if (e0 + r1 < E) {
          float f1 = d1[j] + bias2;
          float v1 = feat[(size_t)sS[r1] * HD + c] * f1 * sCoef[r1 * HH + h];
          unsafeAtomicAdd(&agg[(size_t)sD[r1] * HD + c], v1);
        }
      }
    }
  }
}

// ---------------- Kernel 6: out = ssp(ssp(agg) @ mw1 + mb1) @ mw2 + mb2 ----------------
__global__ void __launch_bounds__(128)
mlp_kernel(const float* __restrict__ agg, const float* __restrict__ w1,
           const float* __restrict__ b1, const float* __restrict__ w2,
           const float* __restrict__ b2, float* __restrict__ out, int N) {
  __shared__ __align__(16) u16 sW1[DIMN * HD];   // mlp_w1^T [64][128] 16KB
  __shared__ __align__(16) u16 sW2[DIMN * DIMN]; // mlp_w2^T [64][64]   8KB
  __shared__ float sB1[DIMN], sB2[DIMN];
  __shared__ __align__(16) u16 sA[32 * HD];      // 8KB
  __shared__ __align__(16) u16 sH[32 * DIMN];    // 4KB
  for (int idx = threadIdx.x; idx < HD * DIMN; idx += blockDim.x) {
    int k = idx / DIMN, n = idx % DIMN;
    sW1[n * HD + k] = f2bf(w1[idx]);
  }
  for (int idx = threadIdx.x; idx < DIMN * DIMN; idx += blockDim.x) {
    int k = idx / DIMN, n = idx % DIMN;
    sW2[n * DIMN + k] = f2bf(w2[idx]);
  }
  for (int idx = threadIdx.x; idx < DIMN; idx += blockDim.x) {
    sB1[idx] = b1[idx];
    sB2[idx] = b2[idx];
  }
  int nt = (N + 31) / 32;
  int lane = threadIdx.x & 31, wv = threadIdx.x >> 5, half = lane >> 4;
  for (int t = blockIdx.x; t < nt; t += gridDim.x) {
    int n0 = t * 32;
    bool full = (n0 + 32 <= N);
    __syncthreads();
    if (full) {
      for (int idx = threadIdx.x; idx < 32 * HD; idx += blockDim.x)
        sA[idx] = f2bf(sspf(agg[(size_t)n0 * HD + idx]));
    } else {
      for (int idx = threadIdx.x; idx < 32 * HD; idx += blockDim.x) {
        int r = idx / HD, c = idx % HD;
        int node = n0 + r;
        sA[idx] = f2bf(node < N ? sspf(agg[(size_t)node * HD + c]) : 0.f);
      }
    }
    __syncthreads();
    v8f acc0 = {}, acc1 = {};
#pragma unroll
    for (int kb = 0; kb < HD; kb += 32) {
      Frag b = loadB(sW1, HD, wv * 16, kb);
      Frag a0 = loadA(sA, HD, kb);
      Frag a1 = loadA(sA + 16 * HD, HD, kb);
      acc0 = bfwmma(a0, b, acc0);
      acc1 = bfwmma(a1, b, acc1);
    }
    int c1 = wv * 16 + (lane & 15);
    float bias1 = sB1[c1];
    {
      u16* q = sH + (8 * half) * DIMN + c1;
#pragma unroll
      for (int j = 0; j < 8; j++) {
        q[j * DIMN] = f2bf(sspf(acc0[j] + bias1));
        q[(j + 16) * DIMN] = f2bf(sspf(acc1[j] + bias1));
      }
    }
    __syncthreads();
    v8f d0 = {}, d1 = {};
#pragma unroll
    for (int kb = 0; kb < DIMN; kb += 32) {
      Frag b = loadB(sW2, DIMN, wv * 16, kb);
      Frag a0 = loadA(sH, DIMN, kb);
      Frag a1 = loadA(sH + 16 * DIMN, DIMN, kb);
      d0 = bfwmma(a0, b, d0);
      d1 = bfwmma(a1, b, d1);
    }
    int c = wv * 16 + (lane & 15);
    float bias2 = sB2[c];
    if (full) {
      // one base address per lane; rows become immediate byte offsets (stride 256B)
      float* p0 = out + (size_t)(n0 + 8 * half) * DIMN + c;
#pragma unroll
      for (int j = 0; j < 8; j++) {
        p0[j * DIMN] = d0[j] + bias2;
        p0[(j + 16) * DIMN] = d1[j] + bias2;
      }
    } else {
#pragma unroll
      for (int j = 0; j < 8; j++) {
        int r = j + 8 * half;
        if (n0 + r < N) out[(size_t)(n0 + r) * DIMN + c] = d0[j] + bias2;
        if (n0 + r + 16 < N) out[(size_t)(n0 + r + 16) * DIMN + c] = d1[j] + bias2;
      }
    }
  }
}

extern "C" void kernel_launch(void* const* d_in, const int* in_sizes, int n_in,
                              void* d_out, int out_size, void* d_ws, size_t ws_size,
                              hipStream_t stream) {
  const float* x   = (const float*)d_in[0];
  const float* gs  = (const float*)d_in[1];
  const float* rij = (const float*)d_in[2];
  const int*   src = (const int*)d_in[3];
  const int*   dst = (const int*)d_in[4];
  const float* fcw = (const float*)d_in[5];
  const float* al  = (const float*)d_in[6];
  const float* ar  = (const float*)d_in[7];
  const float* fw1 = (const float*)d_in[8];
  const float* fb1 = (const float*)d_in[9];
  const float* fw2 = (const float*)d_in[10];
  const float* fb2 = (const float*)d_in[11];
  const float* mw1 = (const float*)d_in[12];
  const float* mb1 = (const float*)d_in[13];
  const float* mw2 = (const float*)d_in[14];
  const float* mb2 = (const float*)d_in[15];
  (void)n_in; (void)out_size; (void)ws_size;

  int N = in_sizes[0] / DIMN;
  int E = in_sizes[2];
  float* out = (float*)d_out;

  char* p = (char*)d_ws;
  float* feat = (float*)p; p += (size_t)N * HD * 4;
  float* el   = (float*)p; p += (size_t)N * HH * 4;
  float* er   = (float*)p; p += (size_t)N * HH * 4;
  u32*   mkey = (u32*)p;   p += (size_t)N * HH * 4;
  float* den  = (float*)p; p += (size_t)N * HH * 4;
  float* esc  = (float*)p; p += (size_t)E * HH * 4;
  float* agg  = (float*)p; p += (size_t)N * HD * 4;

  hipMemsetAsync(mkey, 0, (size_t)N * HH * 4, stream);  // key 0 -> NaN -> treated as 0
  hipMemsetAsync(den, 0, (size_t)N * HH * 4, stream);
  hipMemsetAsync(agg, 0, (size_t)N * HD * 4, stream);

  int ntN = (N + 31) / 32;
  int ntE = (E + 31) / 32;
  int gFeat = ntN < 1024 ? ntN : 1024;
  int gMsg  = ntE < 2048 ? ntE : 2048;
  int gMlp  = ntN < 1024 ? ntN : 1024;

  feat_kernel<<<dim3(gFeat), dim3(256), 0, stream>>>(x, fcw, feat, N);
  attn_kernel<<<dim3((N + 7) / 8), dim3(256), 0, stream>>>(feat, al, ar, el, er, N);
  score_kernel<<<dim3((E + 255) / 256), dim3(256), 0, stream>>>(src, dst, el, er, esc, mkey, E);
  exp_kernel<<<dim3((E + 255) / 256), dim3(256), 0, stream>>>(dst, esc, mkey, den, E);
  msg_kernel<<<dim3(gMsg), dim3(256), 0, stream>>>(gs, rij, src, dst, fw1, fb1,
                                                   fw2, fb2, feat, esc, den, agg, E);
  mlp_kernel<<<dim3(gMlp), dim3(128), 0, stream>>>(agg, mw1, mb1, mw2, mb2, out, N);
}
